// GraphSmote_53472342835646
// MI455X (gfx1250) — matrix operand order repaired and checked by reading, not compiled, not run on hardware
//
#include <hip/hip_runtime.h>
#include <hip/hip_bf16.h>

typedef __attribute__((ext_vector_type(16))) _Float16 v16h;
typedef __attribute__((ext_vector_type(8)))  _Float16 v8h;
typedef __attribute__((ext_vector_type(8)))  float    v8f;

#define D_DIM 256
// LDS tile: 16 rows x 256 halves, padded row pitch (in halves) to spread banks
#define TILE_PITCH_H 264              // 528 bytes per row (stride of 4 banks)
#define TILE_H (16 * TILE_PITCH_H)    // halves per buffer (8448 B)

__device__ __forceinline__ unsigned hash_u32(unsigned s) {
  s ^= s >> 16; s *= 0x7feb352du;
  s ^= s >> 15; s *= 0x846ca68bu;
  s ^= s >> 16; return s;
}

// ---------------------------------------------------------------------------
// 1) Order-preserving partition of idx_train into chosen (y==1) / benign (y==0)
// ---------------------------------------------------------------------------
__global__ void partition_kernel(const int* __restrict__ idx_train,
                                 const int* __restrict__ y,
                                 int n_train,
                                 int* __restrict__ chosen,
                                 int* __restrict__ benign) {
  int lane = threadIdx.x;
  int cc = 0, bc = 0;
  for (int base = 0; base < n_train; base += 32) {
    int i = base + lane;
    int idx = 0, lab = -1;
    if (i < n_train) { idx = idx_train[i]; lab = y[idx]; }
    unsigned long long mc = __ballot(lab == 1);
    unsigned long long mb = __ballot(lab == 0);
    unsigned long long lt = (1ull << lane) - 1ull;
    if (lab == 1) chosen[cc + __popcll(mc & lt)] = idx;
    if (lab == 0) benign[bc + __popcll(mb & lt)] = idx;
    cc += (int)__popcll(mc);
    bc += (int)__popcll(mb);
  }
}

// ---------------------------------------------------------------------------
// 2) Deterministic stand-in for jax.random sampling of benign rows
// ---------------------------------------------------------------------------
__global__ void sample_kernel(const int* __restrict__ benign, int m,
                              int* __restrict__ sampledIdx, int n, int n_pad) {
  int i = blockIdx.x * blockDim.x + threadIdx.x;
  if (i >= n_pad) return;
  int v = 0;
  if (i < n && m > 0) v = benign[hash_u32((unsigned)i ^ 0x2au) % (unsigned)m];
  sampledIdx[i] = v;
}

// ---------------------------------------------------------------------------
// 3) Gather rows of x -> contiguous f16 matrix + fp32 squared row norms.
// ---------------------------------------------------------------------------
__global__ void gather_f16_kernel(const float* __restrict__ x,
                                  const int* __restrict__ rows,
                                  _Float16* __restrict__ dst,
                                  float* __restrict__ norms,
                                  int n, float pad_norm) {
  __shared__ float red[256];
  int i = blockIdx.x;
  int t = threadIdx.x;
  float v = 0.0f;
  if (i < n) v = x[(size_t)rows[i] * D_DIM + t];
  dst[(size_t)i * D_DIM + t] = (_Float16)v;
  red[t] = v * v;
  __syncthreads();
  for (int s = 128; s > 0; s >>= 1) {
    if (t < s) red[t] += red[t + s];
    __syncthreads();
  }
  if (t == 0) norms[i] = (i < n) ? red[0] : pad_norm;
}

// ---------------------------------------------------------------------------
// Async global->LDS copy of one 16x256 f16 B tile (8 KB), ASYNCcnt-tracked.
// Each thread issues 2 x global_load_async_to_lds_b128 (GVS mode: SGPR base +
// per-lane byte offset; VDST supplies the per-lane LDS byte address).
// ---------------------------------------------------------------------------
__device__ __forceinline__ void async_b_tile(const _Float16* __restrict__ Bf,
                                             unsigned lds_base_bytes,
                                             int jt, int tid) {
#pragma unroll
  for (int s = 0; s < 2; ++s) {
    int c = tid + s * 256;            // chunk id in [0,512): 16B chunks
    int row = c >> 5;                 // 0..15
    int k16 = c & 31;                 // 16B chunk within the 512B row
    unsigned goff = (unsigned)((((jt + row) * D_DIM) + k16 * 8) * 2);
    unsigned loff = lds_base_bytes + (unsigned)(row * (TILE_PITCH_H * 2) + k16 * 16);
    asm volatile("global_load_async_to_lds_b128 %0, %1, %2"
                 :
                 : "v"(loff), "v"(goff),
                   "s"((unsigned long long)(uintptr_t)Bf)
                 : "memory");
  }
}

// ---------------------------------------------------------------------------
// 4) Fused distance^2 + row argmin via v_wmma_f32_16x16x32_f16.
//    A fragment (16x256) resident in VGPRs per wave; B tiles double-buffered
//    in LDS via async copies, shared by all 8 waves of the block.
// ---------------------------------------------------------------------------
__global__ __launch_bounds__(256) void dist_argmin_kernel(
    const _Float16* __restrict__ Af, const _Float16* __restrict__ Bf,
    const float* __restrict__ an, const float* __restrict__ bn,
    int* __restrict__ nb, int n, int n_pad) {
  __shared__ _Float16 smem[2 * TILE_H];

  const int tid  = threadIdx.x;
  const int lane = tid & 31;
  const int wave = tid >> 5;
  const int row0 = blockIdx.x * 128 + wave * 16;
  const int mrow = lane & 15;   // A row / B col / C col within tile
  const int hi   = lane >> 4;   // half-wave select

  const unsigned lds_base = (unsigned)(uintptr_t)(&smem[0]);

  // Preload A fragments: 16-bit A 16x32 layout = two interleaved 8xf16 chunks
  const _Float16* arow = Af + (size_t)(row0 + mrow) * D_DIM;
  v16h a[8];
#pragma unroll
  for (int kb = 0; kb < 8; ++kb) {
    v8h lo = *(const v8h*)(arow + kb * 32 + hi * 8);
    v8h hv = *(const v8h*)(arow + kb * 32 + 16 + hi * 8);
#pragma unroll
    for (int j = 0; j < 8; ++j) { a[kb][j] = lo[j]; a[kb][8 + j] = hv[j]; }
  }

  float anv[8];
#pragma unroll
  for (int r = 0; r < 8; ++r) anv[r] = an[row0 + r + hi * 8];

  float bestv[8]; int bestj[8];
#pragma unroll
  for (int r = 0; r < 8; ++r) { bestv[r] = 3.0e38f; bestj[r] = 0; }

  // Prologue: start async fetch of tile 0 into buffer 0.
  async_b_tile(Bf, lds_base, 0, tid);

  for (int jt = 0; jt < n_pad; jt += 16) {
    const int cur = (jt >> 4) & 1;

    // Prefetch next tile into the other buffer (freed by last iter's barrier),
    // then wait for the current tile (in-order ASYNCcnt completion).
    if (jt + 16 < n_pad) {
      async_b_tile(Bf, lds_base + (cur ^ 1) * (TILE_H * 2), jt + 16, tid);
      asm volatile("s_wait_asynccnt 0x2" ::: "memory");
    } else {
      asm volatile("s_wait_asynccnt 0x0" ::: "memory");
    }
    __syncthreads();   // current tile visible to all waves

    // Preload all 8 B fragments from LDS (B 32x16: same interleaved layout as A,
    // two 8xf16 chunks at K 8h.. and 16+8h..), then run the WMMA chain.
    const _Float16* brow = smem + cur * TILE_H + mrow * TILE_PITCH_H + hi * 8;
    v16h bfr[8];
#pragma unroll
    for (int kb = 0; kb < 8; ++kb) {
      v8h lo = *(const v8h*)(brow + kb * 32);
      v8h hv = *(const v8h*)(brow + kb * 32 + 16);
#pragma unroll
      for (int j = 0; j < 8; ++j) { bfr[kb][j] = lo[j]; bfr[kb][8 + j] = hv[j]; }
    }

    v8f c = {};
#pragma unroll
    for (int kb = 0; kb < 8; ++kb) {
      c = __builtin_amdgcn_wmma_f32_16x16x32_f16(false, a[kb], false, bfr[kb],
                                                 (short)0, c, false, false);
    }

    int col = jt + mrow;
    float bcol = bn[col];
#pragma unroll
    for (int r = 0; r < 8; ++r) {
      int row = row0 + r + hi * 8;
      float d2 = anv[r] + bcol - 2.0f * c[r];
      if (col == row) d2 = 3.4e38f;                 // mask diagonal
      if (d2 < bestv[r]) { bestv[r] = d2; bestj[r] = col; }
    }

    __syncthreads();   // all waves done reading `cur` before it is overwritten
  }

  // Reduce (value, index) across the 16 column-lanes of each half-wave.
#pragma unroll
  for (int r = 0; r < 8; ++r) {
    float v = bestv[r]; int j = bestj[r];
#pragma unroll
    for (int mk = 8; mk >= 1; mk >>= 1) {
      float ov = __shfl_xor(v, mk, 32);
      int   oj = __shfl_xor(j, mk, 32);
      if (ov < v || (ov == v && oj < j)) { v = ov; j = oj; }
    }
    int row = row0 + r + hi * 8;
    if (mrow == 0 && row < n) nb[row] = j;
  }
}

// ---------------------------------------------------------------------------
// 5) Output assembly
// ---------------------------------------------------------------------------
__global__ void copy_x_kernel(const float4* __restrict__ src,
                              float4* __restrict__ dst, long count4) {
  long i = (long)blockIdx.x * blockDim.x + threadIdx.x;
  if (i < count4) dst[i] = src[i];
}

__global__ void interp_kernel(const float* __restrict__ x,
                              const int* __restrict__ chosen,
                              const int* __restrict__ sampledIdx,
                              const int* __restrict__ nb,
                              float* __restrict__ out_embed,
                              int N, float t) {
  int i = blockIdx.x, d = threadIdx.x;
  int rc = chosen[i];
  int rb = sampledIdx[nb[i]];
  float xc = x[(size_t)rc * D_DIM + d];
  float xb = x[(size_t)rb * D_DIM + d];
  out_embed[(size_t)(N + i) * D_DIM + d] = xc + (xb - xc) * t;
}

__global__ void tail_kernel(const int* __restrict__ y,
                            const int* __restrict__ idxt,
                            float* __restrict__ out,
                            long yBase, int N, int NT, int n) {
  long i = (long)blockIdx.x * blockDim.x + threadIdx.x;
  long ylen = (long)N + n;
  long ilen = (long)NT + n;
  if (i < ylen) {
    out[yBase + i] = (i < (long)N) ? (float)y[i] : 1.0f;
  } else if (i < ylen + ilen) {
    long j = i - ylen;
    out[yBase + ylen + j] =
        (j < (long)NT) ? (float)idxt[j] : (float)(N + (int)(j - NT));
  }
}

// ---------------------------------------------------------------------------
extern "C" void kernel_launch(void* const* d_in, const int* in_sizes, int n_in,
                              void* d_out, int out_size, void* d_ws, size_t ws_size,
                              hipStream_t stream) {
  const float* x  = (const float*)d_in[0];
  const int* y    = (const int*)d_in[1];
  const int* idxt = (const int*)d_in[2];
  const int N  = in_sizes[1];
  const int NT = in_sizes[2];
  const int D  = D_DIM;

  // n is data-dependent but fully recoverable from the fixed output size:
  // out_size = (N+n)*D + (N+n) + (NT+n)
  long n_l = ((long)out_size - (long)N * D - (long)N - (long)NT) / (long)(D + 2);
  int n = (int)n_l; if (n < 0) n = 0;
  int m = NT - n;
  int n_pad = (n + 127) & ~127; if (n_pad == 0) n_pad = 128;

  char* ws = (char*)d_ws;
  size_t off = 0;
  auto alloc = [&](size_t bytes) {
    size_t o = off; off = (off + bytes + 255) & ~(size_t)255; return o;
  };
  int*      chosen     = (int*)(ws + alloc((size_t)n_pad * 4));
  int*      benign     = (int*)(ws + alloc((size_t)NT * 4));
  int*      sampledIdx = (int*)(ws + alloc((size_t)n_pad * 4));
  int*      nb         = (int*)(ws + alloc((size_t)n_pad * 4));
  float*    an         = (float*)(ws + alloc((size_t)n_pad * 4));
  float*    bn         = (float*)(ws + alloc((size_t)n_pad * 4));
  _Float16* Af         = (_Float16*)(ws + alloc((size_t)n_pad * D * 2));
  _Float16* Bf         = (_Float16*)(ws + alloc((size_t)n_pad * D * 2));

  partition_kernel<<<1, 32, 0, stream>>>(idxt, y, NT, chosen, benign);
  sample_kernel<<<(n_pad + 255) / 256, 256, 0, stream>>>(benign, m, sampledIdx, n, n_pad);
  gather_f16_kernel<<<n_pad, 256, 0, stream>>>(x, chosen,     Af, an, n, 0.0f);
  gather_f16_kernel<<<n_pad, 256, 0, stream>>>(x, sampledIdx, Bf, bn, n, 1.0e30f);
  dist_argmin_kernel<<<n_pad / 128, 256, 0, stream>>>(Af, Bf, an, bn, nb, n, n_pad);

  long count4 = ((long)N * D) / 4;
  copy_x_kernel<<<(unsigned)((count4 + 255) / 256), 256, 0, stream>>>(
      (const float4*)x, (float4*)d_out, count4);
  if (n > 0) {
    interp_kernel<<<n, 256, 0, stream>>>(x, chosen, sampledIdx, nb,
                                         (float*)d_out, N, 0.3745401188f);
  }
  long yBase = (long)(N + n) * D;
  long tail  = (long)(N + n) + (long)(NT + n);
  tail_kernel<<<(unsigned)((tail + 255) / 256), 256, 0, stream>>>(
      y, idxt, (float*)d_out, yBase, N, NT, n);
}
